// ResGNNBlock_30408368456112
// MI455X (gfx1250) — matrix operand order repaired and checked by reading, not compiled, not run on hardware
//
#include <hip/hip_runtime.h>
#include <hip/hip_bf16.h>
#include <math.h>

// ---------------------------------------------------------------------------
// ResGNNBlock (2x GATv2 + LN + GeLU) for MI455X (gfx1250, wave32, WMMA).
//
//   N=50000 nodes, E=400000 edges, D=64, H=4, C=64, H*C=256, ED=16.
//
// Bandwidth-bound: edge gathers/scatters dominate.  GEMMs use
// v_wmma_f32_16x16x32_bf16 (fp32 accum); activations stored bf16 to halve
// gather traffic; all gathers are 16B vector loads; WMMA fragments are built
// with 16B LDS vector loads (A rows contiguous, W staged transposed).
// ---------------------------------------------------------------------------

#define GN   50000      // nodes
#define GE   400000     // edges
#define GD   64         // model dim
#define GH   4          // heads
#define GC   64         // channels/head
#define GHC  256        // H*C
#define GED  16         // edge feature dim
#define ECH  50000      // edge chunk (8 chunks)

typedef __attribute__((ext_vector_type(8)))  float  v8f;
typedef __attribute__((ext_vector_type(16))) __bf16 v16bf;
typedef __attribute__((ext_vector_type(8)))  __bf16 v8bf;

// ------------------------------- helpers -----------------------------------

__device__ __forceinline__ void atomicMaxF32(float* addr, float val) {
  // ordered-int trick; init value must be -inf (0xFF800000)
  if (val >= 0.0f) atomicMax((int*)addr, __float_as_int(val));
  else             atomicMin((unsigned int*)addr, __float_as_uint(val));
}

__device__ __forceinline__ float gelu_exact(float x) {
  return 0.5f * x * (1.0f + erff(x * 0.70710678118654752f));
}

// unpack a dword holding two bf16 (lo = even channel, hi = odd channel)
__device__ __forceinline__ float2 bfpair(unsigned u) {
  float2 r;
  r.x = __uint_as_float(u << 16);
  r.y = __uint_as_float(u & 0xffff0000u);
  return r;
}

// assemble v16bf fragment from two contiguous 8-element LDS runs
__device__ __forceinline__ v16bf frag(const __bf16* lo, const __bf16* hi) {
  v8bf l = *(const v8bf*)lo;
  v8bf h = *(const v8bf*)hi;
  v16bf f;
  #pragma unroll
  for (int j = 0; j < 8; ++j) { f[j] = l[j]; f[j + 8] = h[j]; }
  return f;
}

// --------------------------- node transform GEMM ----------------------------
// out_bf16[N][256] = bf16( h[N][64] @ W[64][256] + bias[256] )
// grid = (2, N/16); block = 256 (8 waves); wave -> one 16x16 tile; K=64 as
// two v_wmma_f32_16x16x32_bf16 steps.  W staged transposed in LDS so both
// A and B fragments are 16B ds_load vectors (ISA 7.12.2 bf16 layouts).
__global__ __launch_bounds__(256) void k_gemm_node(
    const float* __restrict__ h, const float* __restrict__ W,
    const float* __restrict__ bias, __bf16* __restrict__ out)
{
  __shared__ __bf16 sA[16][80];    // 16x64 h tile  (160B rows, 16B aligned)
  __shared__ __bf16 sB[128][80];   // W^T tile [n_local][k], 64 K used
  const int rowBase  = blockIdx.y * 16;
  const int colBlock = blockIdx.x * 128;
  const int t = threadIdx.x;

  // stage A: 1024 floats, one float4 per thread, coalesced
  {
    const int idx = t * 4;
    const int r = idx >> 6, c = idx & 63;
    const float* gp = h + (size_t)(rowBase + r) * GD + c;
    const float4 v = *(const float4*)gp;
    sA[r][c + 0] = (__bf16)v.x; sA[r][c + 1] = (__bf16)v.y;
    sA[r][c + 2] = (__bf16)v.z; sA[r][c + 3] = (__bf16)v.w;
    // speculative prefetch of the next 16-row tile (global_prefetch_b8)
    __builtin_prefetch(gp + 16 * GD, 0, 1);
  }
  // stage B transposed: 64x128 floats, 8 float4 per thread
  #pragma unroll
  for (int i = 0; i < 8; ++i) {
    const int fidx = (t + i * 256) * 4;       // 0..8191
    const int k = fidx >> 7, c = fidx & 127;
    const float4 v = *(const float4*)(W + (size_t)k * GHC + colBlock + c);
    sB[c + 0][k] = (__bf16)v.x; sB[c + 1][k] = (__bf16)v.y;
    sB[c + 2][k] = (__bf16)v.z; sB[c + 3][k] = (__bf16)v.w;
  }
  __syncthreads();

  const int wave = t >> 5, lane = t & 31;
  const int m  = lane & 15;            // A row / D column-within-tile
  const int hi = lane >> 4;            // half-wave select
  const int nl = wave * 16 + m;        // local output column
  const int n  = colBlock + nl;

  v8f acc = {};
  #pragma unroll
  for (int k0 = 0; k0 < GD; k0 += 32) {
    // A lanes 0-15: K = k0+{0-7,16-23}; lanes 16-31: +8 (ISA 7.12.2)
    v16bf a = frag(&sA[m][k0 + hi * 8], &sA[m][k0 + 16 + hi * 8]);
    // B lanes 0-15: K = k0+{0-7,16-23}; lanes 16-31: +8 (same as A)
    v16bf b = frag(&sB[nl][k0 + hi * 8], &sB[nl][k0 + 16 + hi * 8]);
    acc = __builtin_amdgcn_wmma_f32_16x16x32_bf16(
        false, a, false, b, (short)0, acc, false, false);
  }

  // C/D layout: VGPR r -> M = r + 8*hi, lane&15 -> N
  const float bn = bias[n];
  #pragma unroll
  for (int r = 0; r < 8; ++r) {
    const int mm = rowBase + r + hi * 8;
    out[(size_t)mm * GHC + n] = (__bf16)(acc[r] + bn);
  }
}

// --------------------------- edge transform GEMM ----------------------------
// ez_bf16[CH][256] = bf16( edge_attr[CH][16] @ We[16][256] ), K padded 16->32.
__global__ __launch_bounds__(256) void k_gemm_edge(
    const float* __restrict__ ea, const float* __restrict__ We,
    __bf16* __restrict__ ez, int rowOffset)
{
  __shared__ __bf16 sA[16][48];    // 16x32 (16 data + 16 zero pad), 96B rows
  __shared__ __bf16 sB[128][48];   // We^T [n_local][k], zero pad K=16..31
  const int rowBase  = blockIdx.y * 16;
  const int colBlock = blockIdx.x * 128;
  const int t = threadIdx.x;

  {  // stage A (256 floats) + zero pad
    const int r = t >> 4, c = t & 15;
    sA[r][c]      = (__bf16)ea[(size_t)(rowOffset + rowBase + r) * GED + c];
    sA[r][c + 16] = (__bf16)0.0f;
  }
  // stage B transposed: 16x128 floats, 2 float4 per thread
  #pragma unroll
  for (int i = 0; i < 2; ++i) {
    const int fidx = (t + i * 256) * 4;       // 0..2047
    const int k = fidx >> 7, c = fidx & 127;
    const float4 v = *(const float4*)(We + (size_t)k * GHC + colBlock + c);
    sB[c + 0][k] = (__bf16)v.x; sB[c + 1][k] = (__bf16)v.y;
    sB[c + 2][k] = (__bf16)v.z; sB[c + 3][k] = (__bf16)v.w;
  }
  // zero pad of B: 128 cols x 16 K -> 8 elements per thread
  #pragma unroll
  for (int i = 0; i < 8; ++i) {
    const int idx = t + i * 256;
    sB[idx >> 4][16 + (idx & 15)] = (__bf16)0.0f;
  }
  __syncthreads();

  const int wave = t >> 5, lane = t & 31;
  const int m  = lane & 15;
  const int hi = lane >> 4;
  const int nl = wave * 16 + m;
  const int n  = colBlock + nl;

  v16bf a = frag(&sA[m][hi * 8], &sA[m][16 + hi * 8]);
  v16bf b = frag(&sB[nl][hi * 8], &sB[nl][16 + hi * 8]);
  v8f acc = {};
  acc = __builtin_amdgcn_wmma_f32_16x16x32_bf16(
      false, a, false, b, (short)0, acc, false, false);

  #pragma unroll
  for (int r = 0; r < 8; ++r) {
    const int mm = rowBase + r + hi * 8;
    ez[(size_t)mm * GHC + n] = (__bf16)acc[r];
  }
}

// ----------------------------- softmax pieces -------------------------------

__global__ void k_init_seg(float* __restrict__ amax, float* __restrict__ denom,
                           int n) {
  const int i = blockIdx.x * blockDim.x + threadIdx.x;
  if (i < n) { amax[i] = __int_as_float(0xFF800000); denom[i] = 0.0f; }
}

// attention logits per (edge, head) + running segment max at dst.
// all gathers are 16B uint4 loads (8 bf16 each).
__global__ void k_alpha(const int* __restrict__ src, const int* __restrict__ dst,
                        const __bf16* __restrict__ xl, const __bf16* __restrict__ xr,
                        const __bf16* __restrict__ ez, const float* __restrict__ att,
                        float* __restrict__ alpha, float* __restrict__ amax,
                        int e0, int ecnt)
{
  const int idx = blockIdx.x * blockDim.x + threadIdx.x;
  if (idx >= ecnt * GH) return;
  const int el = idx >> 2, hd = idx & 3;
  const int e = e0 + el;
  const int s = src[e], d = dst[e];
  const uint4*  pl = (const uint4*)(xl + (size_t)s  * GHC + hd * GC);
  const uint4*  pr = (const uint4*)(xr + (size_t)d  * GHC + hd * GC);
  const uint4*  pe = (const uint4*)(ez + (size_t)el * GHC + hd * GC);
  const float2* pa = (const float2*)(att + hd * GC);
  float acc = 0.0f;
  #pragma unroll 2
  for (int g = 0; g < 8; ++g) {                 // 8 channels per iteration
    const uint4 vl = pl[g], vr = pr[g], ve = pe[g];
    const unsigned ul[4] = {vl.x, vl.y, vl.z, vl.w};
    const unsigned ur[4] = {vr.x, vr.y, vr.z, vr.w};
    const unsigned ue[4] = {ve.x, ve.y, ve.z, ve.w};
    #pragma unroll
    for (int q = 0; q < 4; ++q) {
      const float2 fl = bfpair(ul[q]);
      const float2 fr = bfpair(ur[q]);
      const float2 fe = bfpair(ue[q]);
      const float2 aw = pa[g * 4 + q];
      float x0 = fl.x + fr.x + fe.x; x0 = (x0 > 0.0f) ? x0 : 0.2f * x0;
      float x1 = fl.y + fr.y + fe.y; x1 = (x1 > 0.0f) ? x1 : 0.2f * x1;
      acc += x0 * aw.x + x1 * aw.y;
    }
  }
  alpha[(size_t)e * GH + hd] = acc;
  atomicMaxF32(&amax[(size_t)d * GH + hd], acc);
}

__global__ void k_expdenom(const int* __restrict__ dst,
                           const float* __restrict__ alpha,
                           const float* __restrict__ amax,
                           float* __restrict__ ex, float* __restrict__ denom)
{
  const int idx = blockIdx.x * blockDim.x + threadIdx.x;
  if (idx >= GE * GH) return;
  const int e = idx >> 2, hd = idx & 3;
  const int d = dst[e];
  float am = amax[(size_t)d * GH + hd];
  if (am < -1e38f) am = 0.0f;                   // where(isfinite, amax, 0)
  const float v = __expf(alpha[idx] - am);
  ex[idx] = v;
  atomicAdd(&denom[(size_t)d * GH + hd], v);
}

// scatter-add of xl[src] * softmax weight into agg[N][H][C]
__global__ void k_agg(const int* __restrict__ src, const int* __restrict__ dst,
                      const __bf16* __restrict__ xl, const float* __restrict__ ex,
                      const float* __restrict__ denom, float* __restrict__ agg)
{
  const int idx = blockIdx.x * blockDim.x + threadIdx.x;
  if (idx >= GE * GH) return;
  const int e = idx >> 2, hd = idx & 3;
  const int s = src[e], d = dst[e];
  const float a = ex[idx] / (denom[(size_t)d * GH + hd] + 1e-16f);
  const uint4* pl = (const uint4*)(xl + (size_t)s * GHC + hd * GC);
  float* po = agg + (size_t)d * GHC + hd * GC;
  #pragma unroll 2
  for (int g = 0; g < 8; ++g) {
    const uint4 v = pl[g];
    const unsigned uu[4] = {v.x, v.y, v.z, v.w};
    #pragma unroll
    for (int q = 0; q < 4; ++q) {
      const float2 f = bfpair(uu[q]);
      atomicAdd(&po[g * 8 + q * 2 + 0], f.x * a);
      atomicAdd(&po[g * 8 + q * 2 + 1], f.y * a);
    }
  }
}

// -------- epilogue: head mean + bias, LayerNorm, (residual), GeLU -----------
// one wave32 per node, 2 channels per lane, __shfl_xor reductions
__global__ __launch_bounds__(256) void k_epilogue(
    const float* __restrict__ agg, const float* __restrict__ gbias,
    const float* __restrict__ lng, const float* __restrict__ lnb,
    const float* __restrict__ identity, float* __restrict__ out, int mode)
{
  const int node = blockIdx.x * 8 + (threadIdx.x >> 5);
  const int lane = threadIdx.x & 31;
  if (node >= GN) return;
  const float2* p2 = (const float2*)(agg + (size_t)node * GHC);
  const float2 q0 = p2[lane], q1 = p2[32 + lane], q2 = p2[64 + lane], q3 = p2[96 + lane];
  const float2 gb = ((const float2*)gbias)[lane];
  float x0 = 0.25f * (q0.x + q1.x + q2.x + q3.x) + gb.x;
  float x1 = 0.25f * (q0.y + q1.y + q2.y + q3.y) + gb.y;

  float s = x0 + x1;
  #pragma unroll
  for (int o = 16; o > 0; o >>= 1) s += __shfl_xor(s, o, 32);
  const float mu = s * (1.0f / 64.0f);
  const float d0 = x0 - mu, d1 = x1 - mu;
  float v = d0 * d0 + d1 * d1;
  #pragma unroll
  for (int o = 16; o > 0; o >>= 1) v += __shfl_xor(v, o, 32);
  const float rstd = rsqrtf(v * (1.0f / 64.0f) + 1e-5f);

  const float2 g2 = ((const float2*)lng)[lane];
  const float2 b2 = ((const float2*)lnb)[lane];
  float y0 = d0 * rstd * g2.x + b2.x;
  float y1 = d1 * rstd * g2.y + b2.y;
  if (mode == 1) {
    const float2 id = ((const float2*)(identity + (size_t)node * GD))[lane];
    y0 += id.x; y1 += id.y;
  }
  float2 r; r.x = gelu_exact(y0); r.y = gelu_exact(y1);
  ((float2*)(out + (size_t)node * GD))[lane] = r;
}

// ------------------------------- launcher -----------------------------------

extern "C" void kernel_launch(void* const* d_in, const int* in_sizes, int n_in,
                              void* d_out, int out_size, void* d_ws, size_t ws_size,
                              hipStream_t stream) {
  (void)in_sizes; (void)n_in; (void)out_size; (void)ws_size;

  const float* h     = (const float*)d_in[0];
  const int*   eidx  = (const int*)d_in[1];
  const float* eattr = (const float*)d_in[2];
  const int* src = eidx;            // edge_index[0]
  const int* dst = eidx + GE;       // edge_index[1]

  // per-layer weights: Wl, bl, Wr, br, We, att, bias, ln_g, ln_b
  const float* W[2][9];
  for (int l = 0; l < 2; ++l)
    for (int k = 0; k < 9; ++k)
      W[l][k] = (const float*)d_in[3 + l * 9 + k];

  // workspace carve-out (~155 MB)
  char* ws = (char*)d_ws;
  size_t off = 0;
  auto carve = [&](size_t bytes) -> void* {
    void* p = ws + off; off = (off + bytes + 255) & ~(size_t)255; return p;
  };
  __bf16* xl    = (__bf16*)carve((size_t)GN * GHC * 2);
  __bf16* xr    = (__bf16*)carve((size_t)GN * GHC * 2);
  __bf16* ez    = (__bf16*)carve((size_t)ECH * GHC * 2);
  float*  alpha = (float*) carve((size_t)GE * GH * 4);
  float*  ex    = (float*) carve((size_t)GE * GH * 4);
  float*  amax  = (float*) carve((size_t)GN * GH * 4);
  float*  denom = (float*) carve((size_t)GN * GH * 4);
  float*  agg   = (float*) carve((size_t)GN * GHC * 4);
  float*  h2    = (float*) carve((size_t)GN * GD * 4);

  const dim3 gGemmNode(2, GN / 16);   // 2 col-blocks x 3125 row tiles
  const dim3 gGemmEdge(2, ECH / 16);
  const int  bEH = (ECH * GH + 255) / 256;
  const int  bE  = (GE * GH + 255) / 256;

  for (int layer = 0; layer < 2; ++layer) {
    const float* hin  = (layer == 0) ? h : h2;
    const float* Wl   = W[layer][0]; const float* bl  = W[layer][1];
    const float* Wr   = W[layer][2]; const float* br  = W[layer][3];
    const float* We   = W[layer][4]; const float* att = W[layer][5];
    const float* gb   = W[layer][6];
    const float* lng  = W[layer][7]; const float* lnb = W[layer][8];
    float* outp = (layer == 0) ? h2 : (float*)d_out;

    k_gemm_node<<<gGemmNode, 256, 0, stream>>>(hin, Wl, bl, xl);
    k_gemm_node<<<gGemmNode, 256, 0, stream>>>(hin, Wr, br, xr);
    k_init_seg<<<(GN * GH + 255) / 256, 256, 0, stream>>>(amax, denom, GN * GH);

    for (int ch = 0; ch < GE / ECH; ++ch) {
      k_gemm_edge<<<gGemmEdge, 256, 0, stream>>>(eattr, We, ez, ch * ECH);
      k_alpha<<<bEH, 256, 0, stream>>>(src, dst, xl, xr, ez, att,
                                       alpha, amax, ch * ECH, ECH);
    }

    k_expdenom<<<bE, 256, 0, stream>>>(dst, alpha, amax, ex, denom);
    hipMemsetAsync(agg, 0, (size_t)GN * GHC * 4, stream);
    k_agg<<<bE, 256, 0, stream>>>(src, dst, xl, ex, denom, agg);
    k_epilogue<<<GN / 8, 256, 0, stream>>>(agg, gb, lng, lnb, h, outp, layer);
  }
}
